// SpectralConv_2250562863747
// MI455X (gfx1250) — matrix operand order repaired and checked by reading, not compiled, not run on hardware
//
#include <hip/hip_runtime.h>
#include <stdint.h>

// ---------------------------------------------------------------------------
// SpectralConv on MI455X (gfx1250).
// Split-precision bf16 WMMA (hi + lo residual, 3x v_wmma_f32_16x16x32_bf16 per
// tile-pair) for the two big transforms. LDS holds PRE-PACKED fragment planes
// (hi-plane / lo-plane, one u32 = bf16(K)|bf16(K+1)) so the inner loop builds
// fragments with pure ds_load_b128 and zero VALU repacking.
// ---------------------------------------------------------------------------

typedef __attribute__((ext_vector_type(16))) __bf16 v16bf;
typedef __attribute__((ext_vector_type(8)))  float  v8f;

union FragB { uint32_t u[8]; uint4 q[2]; v16bf v; };

#define PP 20   // plane row pitch in u32 (16 data words + 4 pad): conflict-free,
                // keeps all fragment chunks 16B-aligned for ds_load_b128.

// Split two fp32 (consecutive K values) into one packed hi word and one packed
// lo word: low half = element K, high half = element K+1 (bf16, truncated).
__device__ __forceinline__ void split2(float fa, float fb,
                                       uint32_t& whi, uint32_t& wlo) {
    uint32_t ua = __float_as_uint(fa), ub = __float_as_uint(fb);
    uint32_t ha = ua & 0xFFFF0000u,  hb = ub & 0xFFFF0000u;
    uint32_t la = __float_as_uint(fa - __uint_as_float(ha));
    uint32_t lb = __float_as_uint(fb - __uint_as_float(hb));
    whi = (ha >> 16) | hb;
    wlo = (la >> 16) | (lb & 0xFFFF0000u);
}

// Natural packing: float4 = 4 consecutive K in one row -> words j2, j2+1.
__device__ __forceinline__ void split_quad_row(float4 f, uint32_t* rh,
                                               uint32_t* rl, int j2) {
    uint32_t wh, wl;
    split2(f.x, f.y, wh, wl); rh[j2]     = wh; rl[j2]     = wl;
    split2(f.z, f.w, wh, wl); rh[j2 + 1] = wh; rl[j2 + 1] = wl;
}

// Cross packing (transposed tiles): f0/f1 are the same 4 columns of two
// adjacent K rows (K, K+1); element e goes to LDS row (+e), word j.
__device__ __forceinline__ void split_quad_rows(float4 f0, float4 f1,
                                                uint32_t* bh, uint32_t* bl, int j) {
    uint32_t wh, wl;
    split2(f0.x, f1.x, wh, wl); bh[0 * PP + j] = wh; bl[0 * PP + j] = wl;
    split2(f0.y, f1.y, wh, wl); bh[1 * PP + j] = wh; bl[1 * PP + j] = wl;
    split2(f0.z, f1.z, wh, wl); bh[2 * PP + j] = wh; bl[2 * PP + j] = wl;
    split2(f0.w, f1.w, wh, wl); bh[3 * PP + j] = wh; bl[3 * PP + j] = wl;
}

// A fragment (16-bit A 16x32 layout): words {lh*4+v} and {8+lh*4+v}.
__device__ __forceinline__ void frag_a(const uint32_t* plane, int row, int lh,
                                       FragB& f) {
    const uint32_t* p = plane + row * PP;
    f.q[0] = *(const uint4*)(p + lh * 4);
    f.q[1] = *(const uint4*)(p + 8 + lh * 4);
}

// B fragment (16-bit B 32x16 layout): words {lh*4+v} and {8+lh*4+v}.
__device__ __forceinline__ void frag_b(const uint32_t* plane, int n, int lh,
                                       FragB& f) {
    const uint32_t* p = plane + n * PP;
    f.q[0] = *(const uint4*)(p + lh * 4);
    f.q[1] = *(const uint4*)(p + 8 + lh * 4);
}

// D += A*B with split precision: hi*hi + hi*lo + lo*hi (lo*lo negligible).
__device__ __forceinline__ v8f wmma3(const FragB& ah, const FragB& al,
                                     const FragB& bh, const FragB& bl, v8f c) {
    c = __builtin_amdgcn_wmma_f32_16x16x32_bf16(false, ah.v, false, bh.v, (short)0, c, false, false);
    c = __builtin_amdgcn_wmma_f32_16x16x32_bf16(false, ah.v, false, bl.v, (short)0, c, false, false);
    c = __builtin_amdgcn_wmma_f32_16x16x32_bf16(false, al.v, false, bh.v, (short)0, c, false, false);
    return c;
}

// ---------------------------------------------------------------------------
// Kernel 0: zero the atomic-accumulation scratch (Hc, Hs, H0).
// ---------------------------------------------------------------------------
__global__ void k_zero(float* __restrict__ p, int n) {
    int i = blockIdx.x * 256 + threadIdx.x;
    if (i < n) p[i] = 0.0f;
}

// ---------------------------------------------------------------------------
// Kernel 1: forward transform (split-K over X, f32-atomic reduction).
//   Hc[b,i,kw] += x[b,i,xk]*wbases_c[b,xk,kw]; Hs likewise (raw sign);
//   H0[b,i,w]  += x[b,i,xk]*wbases_0[b,xk,w]  (VALU, 8 cols).
// Grid: B*64 chunks of K=128. Block: 8 waves = 4 I-tiles x {cos,sin}.
// ---------------------------------------------------------------------------
__global__ __launch_bounds__(256)
void k_fwd(const float* __restrict__ x,   const float* __restrict__ wbc,
           const float* __restrict__ wbs, const float* __restrict__ wb0,
           float* __restrict__ Hc, float* __restrict__ Hs, float* __restrict__ H0) {
    __shared__ uint32_t sAh[64 * PP],   sAl[64 * PP];    // x tile   [i][K=xk]
    __shared__ uint32_t sBch[256 * PP], sBcl[256 * PP];  // wbases_c [kw][K=xk]
    __shared__ uint32_t sBsh[256 * PP], sBsl[256 * PP];  // wbases_s [kw][K=xk]
    __shared__ float    sW0[32 * 8];                     // wbases_0 [xk][w]

    const int b     = blockIdx.x & 3;
    const int chunk = blockIdx.x >> 2;
    const int tid  = threadIdx.x;
    const int lane = tid & 31, wave = tid >> 5;
    const int mt = wave & 3, sel = wave >> 2;
    const int lm = lane & 15, lh = lane >> 4;

    v8f acc[16];
#pragma unroll
    for (int n = 0; n < 16; ++n) acc[n] = (v8f){0,0,0,0,0,0,0,0};
    float h0acc[2] = {0.0f, 0.0f};

    for (int ks = 0; ks < 4; ++ks) {
        const int xk0 = chunk * 128 + ks * 32;

        // ---- stage x tile (K contiguous in memory -> natural packing) ----
        for (int c = tid; c < 512; c += 256) {
            int row = c >> 3, c4 = c & 7;
            const float4 f = *(const float4*)(x + (b * 64 + row) * 8192 + xk0 + c4 * 4);
            split_quad_row(f, &sAh[row * PP], &sAl[row * PP], c4 * 2);
        }
        // ---- stage wbases tiles (K = x is the row dim in memory): load the
        //      float4s of two adjacent x rows and cross-pack ----
        for (int t = tid; t < 1024; t += 256) {
            int xp  = t >> 6;            // x pair index 0..15 -> K word j
            int kw4 = (t & 63) * 4;      // kw row base
            int gi  = (b * 8192 + xk0 + 2 * xp) * 256 + kw4;
            float4 f0 = *(const float4*)(wbc + gi);
            float4 f1 = *(const float4*)(wbc + gi + 256);
            split_quad_rows(f0, f1, &sBch[kw4 * PP], &sBcl[kw4 * PP], xp);
            float4 g0 = *(const float4*)(wbs + gi);
            float4 g1 = *(const float4*)(wbs + gi + 256);
            split_quad_rows(g0, g1, &sBsh[kw4 * PP], &sBsl[kw4 * PP], xp);
        }
        // ---- stage wbases_0 tile: 32 x 8 ----
        {
            int row = tid >> 3, w = tid & 7;
            sW0[row * 8 + w] = wb0[(b * 8192 + xk0 + row) * 8 + w];
        }
        __syncthreads();

        FragB Ah, Al;
        frag_a(sAh, mt * 16 + lm, lh, Ah);
        frag_a(sAl, mt * 16 + lm, lh, Al);
        const uint32_t* Bph = sel ? sBsh : sBch;
        const uint32_t* Bpl = sel ? sBsl : sBcl;
#pragma unroll
        for (int nt = 0; nt < 16; ++nt) {
            FragB Bh, Bl;
            frag_b(Bph, nt * 16 + lm, lh, Bh);
            frag_b(Bpl, nt * 16 + lm, lh, Bl);
            acc[nt] = wmma3(Ah, Al, Bh, Bl, acc[nt]);
        }
        // ---- 0-term (8 columns): VALU, 2 (i,w) pairs per thread ----
#pragma unroll
        for (int q = 0; q < 2; ++q) {
            int p = tid * 2 + q, i = p >> 3, w = p & 7;
            float s = 0.0f;
#pragma unroll
            for (int j = 0; j < 16; ++j) {
                uint32_t whi = sAh[i * PP + j], wlo = sAl[i * PP + j];
                float x0 = __uint_as_float(whi << 16) + __uint_as_float(wlo << 16);
                float x1 = __uint_as_float(whi & 0xFFFF0000u) +
                           __uint_as_float(wlo & 0xFFFF0000u);
                s += x0 * sW0[(2 * j) * 8 + w] + x1 * sW0[(2 * j + 1) * 8 + w];
            }
            h0acc[q] += s;
        }
        __syncthreads();
    }

    // ---- split-K reduction via f32 atomics ----
    float* H = sel ? Hs : Hc;
#pragma unroll
    for (int nt = 0; nt < 16; ++nt)
#pragma unroll
        for (int r = 0; r < 8; ++r) {
            int i  = mt * 16 + r + (lh << 3);
            int kw = nt * 16 + lm;
            atomicAdd(&H[((b << 6) + i) * 256 + kw], acc[nt][r]);
        }
#pragma unroll
    for (int q = 0; q < 2; ++q) {
        int p = tid * 2 + q, i = p >> 3, w = p & 7;
        atomicAdd(&H0[((b << 6) + i) * 8 + w], h0acc[q]);
    }
}

// ---------------------------------------------------------------------------
// Kernel 2: spectral channel mixing (tiny VALU stage; signs and +-2 folded).
//   x_s_hat = -P  (P = raw x@wbases_s)
//   Gc = 2*f_c = 2*(Hc*wc + P*ws); Gs = -2*f_s = -2*(Hc*ws - P*wc); F0 = H0*w0
// ---------------------------------------------------------------------------
__global__ __launch_bounds__(256)
void k_mix(const float* __restrict__ Hc, const float* __restrict__ Hs,
           const float* __restrict__ H0,
           const float* __restrict__ wc, const float* __restrict__ wsw,
           const float* __restrict__ w0,
           float* __restrict__ Gc, float* __restrict__ Gs, float* __restrict__ F0) {
    const int kw = blockIdx.x;
    const int o = threadIdx.x & 63, b = threadIdx.x >> 6;
    float fc = 0.0f, fs = 0.0f;
    for (int i = 0; i < 64; ++i) {
        float hc = Hc[((b << 6) + i) * 256 + kw];
        float p  = Hs[((b << 6) + i) * 256 + kw];
        float a  = wc [((i << 6) + o) * 256 + kw];
        float s  = wsw[((i << 6) + o) * 256 + kw];
        fc += hc * a + p * s;
        fs += hc * s - p * a;
    }
    Gc[((b << 6) + o) * 256 + kw] =  2.0f * fc;
    Gs[((b << 6) + o) * 256 + kw] = -2.0f * fs;
    if (kw < 8) {
        float f0 = 0.0f;
        for (int i = 0; i < 64; ++i)
            f0 += H0[((b << 6) + i) * 8 + kw] * w0[((i << 6) + o) * 8 + kw];
        F0[((b << 6) + o) * 8 + kw] = f0;
    }
}

// ---------------------------------------------------------------------------
// Kernel 3: inverse transform.
//   out[b,o,x] = Gc.bc + Gs.bs (WMMA, K=256 each) + F0.b0 (VALU, 8 cols)
// Grid: B*64 X-tiles of 128. Block: 8 waves = 4 O-tiles x 2 X-halves.
// ---------------------------------------------------------------------------
__global__ __launch_bounds__(256)
void k_inv(const float* __restrict__ Gc, const float* __restrict__ Gs,
           const float* __restrict__ F0,
           const float* __restrict__ bc, const float* __restrict__ bs,
           const float* __restrict__ b0, float* __restrict__ out) {
    __shared__ uint32_t sAch[64 * PP],  sAcl[64 * PP];   // coef  [o][K=kw]
    __shared__ uint32_t sAsh[64 * PP],  sAsl[64 * PP];
    __shared__ uint32_t sBch[128 * PP], sBcl[128 * PP];  // bases [x][K=kw]
    __shared__ uint32_t sBsh[128 * PP], sBsl[128 * PP];
    __shared__ float    sF0[64 * 8],    sB0[128 * 8];

    const int b  = blockIdx.x & 3;
    const int xb = (blockIdx.x >> 2) * 128;
    const int tid  = threadIdx.x;
    const int lane = tid & 31, wave = tid >> 5;
    const int mt = wave & 3, nh = wave >> 2;
    const int lm = lane & 15, lh = lane >> 4;

    for (int p = tid; p < 512;  p += 256) sF0[p] = F0[(b << 9)  + p];
    for (int p = tid; p < 1024; p += 256) sB0[p] = b0[(b << 16) + xb * 8 + p];

    v8f acc[4];
#pragma unroll
    for (int n = 0; n < 4; ++n) acc[n] = (v8f){0,0,0,0,0,0,0,0};

    for (int ks = 0; ks < 8; ++ks) {
        const int kw0 = ks * 32;
        // ---- stage coefficient tiles (K=kw contiguous -> natural packing) ----
        for (int c = tid; c < 512; c += 256) {
            int row = c >> 3, c4 = c & 7;
            int gi  = ((b << 6) + row) * 256 + kw0 + c4 * 4;
            float4 f = *(const float4*)(Gc + gi);
            float4 g = *(const float4*)(Gs + gi);
            split_quad_row(f, &sAch[row * PP], &sAcl[row * PP], c4 * 2);
            split_quad_row(g, &sAsh[row * PP], &sAsl[row * PP], c4 * 2);
        }
        // ---- stage bases tiles (K=kw contiguous -> natural packing) ----
        for (int c = tid; c < 1024; c += 256) {
            int row = c >> 3, c4 = c & 7;
            int gi  = (b * 8192 + xb + row) * 256 + kw0 + c4 * 4;
            float4 f = *(const float4*)(bc + gi);
            float4 g = *(const float4*)(bs + gi);
            split_quad_row(f, &sBch[row * PP], &sBcl[row * PP], c4 * 2);
            split_quad_row(g, &sBsh[row * PP], &sBsl[row * PP], c4 * 2);
        }
        __syncthreads();

        FragB Ach, Acl, Ash, Asl;
        frag_a(sAch, mt * 16 + lm, lh, Ach);
        frag_a(sAcl, mt * 16 + lm, lh, Acl);
        frag_a(sAsh, mt * 16 + lm, lh, Ash);
        frag_a(sAsl, mt * 16 + lm, lh, Asl);
#pragma unroll
        for (int ntl = 0; ntl < 4; ++ntl) {
            int n = (nh * 4 + ntl) * 16 + lm;
            FragB Bh, Bl;
            frag_b(sBch, n, lh, Bh);
            frag_b(sBcl, n, lh, Bl);
            acc[ntl] = wmma3(Ach, Acl, Bh, Bl, acc[ntl]);
            frag_b(sBsh, n, lh, Bh);
            frag_b(sBsl, n, lh, Bl);
            acc[ntl] = wmma3(Ash, Asl, Bh, Bl, acc[ntl]);
        }
        __syncthreads();
    }

    // ---- epilogue: add 0-term and store ----
#pragma unroll
    for (int ntl = 0; ntl < 4; ++ntl)
#pragma unroll
        for (int r = 0; r < 8; ++r) {
            int o  = mt * 16 + r + (lh << 3);
            int xi = (nh * 4 + ntl) * 16 + lm;
            float v = acc[ntl][r];
#pragma unroll
            for (int w = 0; w < 8; ++w) v += sF0[o * 8 + w] * sB0[xi * 8 + w];
            out[((b << 6) + o) * 8192 + xb + xi] = v;
        }
}

// ---------------------------------------------------------------------------
// Launch. Workspace layout (floats):
//   Hc 0 | Hs 65536 | H0 131072 | Gc 133120 | Gs 198656 | F0 264192 (end 266240)
// ---------------------------------------------------------------------------
extern "C" void kernel_launch(void* const* d_in, const int* in_sizes, int n_in,
                              void* d_out, int out_size, void* d_ws, size_t ws_size,
                              hipStream_t stream) {
    const float* x   = (const float*)d_in[0];
    const float* bc  = (const float*)d_in[1];
    const float* bs  = (const float*)d_in[2];
    const float* b0  = (const float*)d_in[3];
    const float* wbc = (const float*)d_in[4];
    const float* wbs = (const float*)d_in[5];
    const float* wb0 = (const float*)d_in[6];
    const float* wc  = (const float*)d_in[7];
    const float* wsw = (const float*)d_in[8];
    const float* w0  = (const float*)d_in[9];
    float* out = (float*)d_out;
    float* ws  = (float*)d_ws;

    float* Hc = ws;
    float* Hs = ws + 65536;
    float* H0 = ws + 131072;
    float* Gc = ws + 133120;
    float* Gs = ws + 198656;
    float* F0 = ws + 264192;

    k_zero<<<(133120 + 255) / 256, 256, 0, stream>>>(ws, 133120);
    k_fwd <<<256, 256, 0, stream>>>(x, wbc, wbs, wb0, Hc, Hs, H0);
    k_mix <<<256, 256, 0, stream>>>(Hc, Hs, H0, wc, wsw, w0, Gc, Gs, F0);
    k_inv <<<256, 256, 0, stream>>>(Gc, Gs, F0, bc, bs, b0, out);
}
